// RoPESelfAttention_11269994185100
// MI455X (gfx1250) — matrix operand order repaired and checked by reading, not compiled, not run on hardware
//
#include <hip/hip_runtime.h>

// ---------------------------------------------------------------------------
// RoPE self-attention for MI455X (gfx1250, wave32, WMMA).
// Pipeline (all bf16 storage, f32 accumulate):
//   1) convert x -> Xbf, Wqkv -> WqkvT(bf16, transposed), Wout -> WoutT
//   2) QKV GEMM (WMMA bf16) + fused RoPE epilogue -> Qbf, Kbf, Vt (V transposed)
//   3) flash attention per (b,h,16-row q tile): WMMA scores, online softmax,
//      P staged through per-wave LDS tile, WMMA P@V -> Attnbf
//   4) out-proj GEMM (WMMA bf16) -> f32 d_out
// ---------------------------------------------------------------------------

typedef __bf16  bf16x16 __attribute__((ext_vector_type(16)));
typedef float   floatx8 __attribute__((ext_vector_type(8)));

constexpr int Bsz  = 2;
constexpr int Lseq = 2048;
constexpr int Dmod = 1024;
constexpr int Hn   = 16;
constexpr int DhC  = 64;
constexpr int Mrows = Bsz * Lseq;     // 4096 flattened rows (weights shared over batch)
constexpr int NQKV  = 3 * Dmod;       // 3072

__device__ __forceinline__ unsigned short f32_to_bf16_bits(float f) {
  unsigned int u = __float_as_uint(f);
  unsigned int r = u + 0x7FFFu + ((u >> 16) & 1u);   // round-to-nearest-even
  return (unsigned short)(r >> 16);
}

union Frag {
  bf16x16 v;
  uint4   q[2];
};

// A-matrix fragment (16x32 bf16). ISA 7.12.2: lane m = lane&15;
// lanes 0-15 hold K 0..7 (v0-3) & 16..23 (v4-7), lanes 16-31 hold K 8..15 & 24..31.
__device__ __forceinline__ Frag load_frag_A(const unsigned short* __restrict__ base,
                                            size_t ld, int row0, int col0, int lane) {
  const int m = lane & 15, hi = lane >> 4;
  const unsigned short* p = base + (size_t)(row0 + m) * ld + col0 + hi * 8;
  Frag f;
  f.q[0] = *(const uint4*)(p);
  f.q[1] = *(const uint4*)(p + 16);
  return f;
}

// B-matrix fragment (32x16 bf16), source stored transposed (N x K row-major):
// lane n = lane&15 reads K 0..7 & 16..23 (lanes 0-15) or K 8..15 & 24..31 (lanes 16-31) of row n.
__device__ __forceinline__ Frag load_frag_B(const unsigned short* __restrict__ baseT,
                                            size_t ld, int n0, int col0, int lane) {
  const int n = lane & 15, hi = lane >> 4;
  const unsigned short* p = baseT + (size_t)(n0 + n) * ld + col0 + hi * 8;
  Frag f;
  f.q[0] = *(const uint4*)(p);
  f.q[1] = *(const uint4*)(p + 16);
  return f;
}

__device__ __forceinline__ floatx8 wmma_bf16(Frag a, Frag b, floatx8 c) {
  return __builtin_amdgcn_wmma_f32_16x16x32_bf16(
      /*neg_a=*/false, a.v, /*neg_b=*/false, b.v,
      /*c_mod=*/(short)0, c, /*reuse_a=*/false, /*reuse_b=*/false);
}

// ---------------------------------------------------------------------------
// Conversion kernels
// ---------------------------------------------------------------------------
__global__ void cvt_f32_bf16(const float* __restrict__ in,
                             unsigned short* __restrict__ out, int n) {
  int i = blockIdx.x * blockDim.x + threadIdx.x;
  if (i < n) out[i] = f32_to_bf16_bits(in[i]);
}

// out[(c,r)] = in[(r,c)] : produce transposed bf16 weights (N x K row-major)
__global__ void cvt_f32_bf16_T(const float* __restrict__ in,
                               unsigned short* __restrict__ out,
                               int rows, int cols) {
  int o = blockIdx.x * blockDim.x + threadIdx.x;
  if (o < rows * cols) {
    int c = o / rows;
    int r = o - c * rows;
    out[o] = f32_to_bf16_bits(in[(size_t)r * cols + c]);
  }
}

// ---------------------------------------------------------------------------
// QKV GEMM + fused RoPE. Each wave: 16(M) x 64(N) output, K-loop over 1024.
// N tiles are head-aligned (Dh=64) so seg/head are uniform per wave and the
// RoPE pair (d, d+32) is accumulator pair (t, t+2) at identical lane/vgpr.
// ---------------------------------------------------------------------------
__global__ __launch_bounds__(256)
void qkv_rope_kernel(const unsigned short* __restrict__ Xbf,
                     const unsigned short* __restrict__ WqkvT,
                     unsigned short* __restrict__ Qbf,
                     unsigned short* __restrict__ Kbf,
                     unsigned short* __restrict__ Vt) {
  const int lane = threadIdx.x & 31;
  const int gw   = blockIdx.x * (blockDim.x >> 5) + (threadIdx.x >> 5);
  const int nW   = NQKV / 64;                 // 48 wave-columns
  const int mt   = gw / nW;
  const int nw   = gw - mt * nW;
  const int row0 = mt * 16;
  const int n0   = nw * 64;

  floatx8 acc[4] = {};
  for (int k0 = 0; k0 < Dmod; k0 += 32) {
    Frag a = load_frag_A(Xbf, Dmod, row0, k0, lane);
    if (k0 + 32 < Dmod) {  // hint next A tile into cache (global_prefetch_b8)
      __builtin_prefetch(Xbf + (size_t)(row0 + (lane & 15)) * Dmod + k0 + 32, 0, 1);
    }
#pragma unroll
    for (int t = 0; t < 4; ++t) {
      Frag b = load_frag_B(WqkvT, Dmod, n0 + 16 * t, k0, lane);
      acc[t] = wmma_bf16(a, b, acc[t]);
    }
  }

  const int n  = lane & 15, hi = lane >> 4;
  const int seg = n0 / Dmod;                  // 0=Q 1=K 2=V
  const int h   = (n0 - seg * Dmod) / DhC;    // head (tile spans exactly one head)

  if (seg < 2) {
    unsigned short* dst = (seg == 0) ? Qbf : Kbf;
#pragma unroll
    for (int j = 0; j < 8; ++j) {
      const int m   = row0 + j + 8 * hi;
      const int pos = m & (Lseq - 1);
#pragma unroll
      for (int t = 0; t < 2; ++t) {
        const int d = 16 * t + n;                       // 0..31
        // inv_freq = 10000^(-d/32) = exp(-d*ln(1e4)/32)
        const float invf = __expf(-(float)d * 0.28782313662f);
        const float ang  = (float)pos * invf;
        float s, c;
        __sincosf(ang, &s, &c);
        const float x1 = acc[t][j], x2 = acc[t + 2][j];
        const size_t base = (size_t)m * Dmod + h * DhC;
        dst[base + d]      = f32_to_bf16_bits(x1 * c - x2 * s);
        dst[base + d + 32] = f32_to_bf16_bits(x1 * s + x2 * c);
      }
    }
  } else {
    // V stored transposed per batch: Vt[b][c][pos], c = h*64+d, row stride L.
#pragma unroll
    for (int j = 0; j < 8; ++j) {
      const int m   = row0 + j + 8 * hi;
      const int b   = m >> 11;              // / Lseq
      const int pos = m & (Lseq - 1);
#pragma unroll
      for (int t = 0; t < 4; ++t) {
        const int c = h * DhC + 16 * t + n;
        Vt[(size_t)b * Dmod * Lseq + (size_t)c * Lseq + pos] =
            f32_to_bf16_bits(acc[t][j]);
      }
    }
  }
}

// ---------------------------------------------------------------------------
// Flash attention: one wave = one (b, h, 16-row q tile); keys in chunks of 32.
// Score C-layout (row = vgpr+8*hi across 16-lane halves) matches per-lane
// stats arrays; row reductions via shfl_xor within each 16-lane group.
// ---------------------------------------------------------------------------
__global__ __launch_bounds__(256)
void attn_kernel(const unsigned short* __restrict__ Qbf,
                 const unsigned short* __restrict__ Kbf,
                 const unsigned short* __restrict__ Vt,
                 const int* __restrict__ mask,
                 unsigned short* __restrict__ Attnbf) {
  __shared__ __align__(16) unsigned short Plds[8][16 * 32];   // 1KB per wave

  const int wib  = threadIdx.x >> 5;
  const int lane = threadIdx.x & 31;
  const int gw   = blockIdx.x * 8 + wib;
  const int qt = gw & 127;            // q tile (L/16 = 128)
  const int h  = (gw >> 7) & 15;
  const int b  = gw >> 11;

  const int n = lane & 15, hi = lane >> 4;
  const unsigned short* VtB = Vt + (size_t)b * Dmod * Lseq;

  // Q A-fragments (16 x 64 split into two 16x32 K-steps), loaded once.
  Frag qf[2];
#pragma unroll
  for (int s = 0; s < 2; ++s)
    qf[s] = load_frag_A(Qbf, Dmod, b * Lseq + qt * 16, h * DhC + s * 32, lane);

  float rmax[8], rsum[8];
#pragma unroll
  for (int j = 0; j < 8; ++j) { rmax[j] = -1e30f; rsum[j] = 0.f; }
  floatx8 O[4] = {};

  for (int k0 = 0; k0 < Lseq; k0 += 32) {
    // ---- scores: S[t] = (q . k^T)/8 for key cols k0+16t+n --------------
    float s2[2][8];
    float mk[2];
#pragma unroll
    for (int t = 0; t < 2; ++t) {
      floatx8 c = {};
#pragma unroll
      for (int s = 0; s < 2; ++s) {
        Frag kf = load_frag_B(Kbf, Dmod, b * Lseq + k0 + 16 * t,
                              h * DhC + s * 32, lane);
        c = wmma_bf16(qf[s], kf, c);
      }
      mk[t] = (mask[b * Lseq + k0 + 16 * t + n] != 0) ? 1.f : 0.f;
#pragma unroll
      for (int j = 0; j < 8; ++j)
        s2[t][j] = (mk[t] != 0.f) ? -1e30f : c[j] * 0.125f;
    }

    // ---- online softmax per row, P into LDS ----------------------------
#pragma unroll
    for (int j = 0; j < 8; ++j) {
      float mx = fmaxf(s2[0][j], s2[1][j]);
#pragma unroll
      for (int off = 1; off < 16; off <<= 1)
        mx = fmaxf(mx, __shfl_xor(mx, off, 32));       // stays in 16-lane half
      const float nm    = fmaxf(rmax[j], mx);
      const float alpha = __expf(rmax[j] - nm);
      rmax[j] = nm;
      const float p0 = __expf(s2[0][j] - nm);
      const float p1 = __expf(s2[1][j] - nm);
      float ls = p0 + p1;
#pragma unroll
      for (int off = 1; off < 16; off <<= 1)
        ls += __shfl_xor(ls, off, 32);
      rsum[j] = rsum[j] * alpha + ls;
#pragma unroll
      for (int t4 = 0; t4 < 4; ++t4) O[t4][j] = O[t4][j] * alpha;
      const int m = j + 8 * hi;                        // C-layout row
      Plds[wib][m * 32 + n]      = f32_to_bf16_bits(p0);
      Plds[wib][m * 32 + 16 + n] = f32_to_bf16_bits(p1);
    }

    // ---- P (16x32, A-layout) from per-wave LDS tile --------------------
    Frag pf;
    {
      const unsigned short* p = &Plds[wib][(lane & 15) * 32 + hi * 8];
      pf.q[0] = *(const uint4*)(p);
      pf.q[1] = *(const uint4*)(p + 16);
    }

    // ---- O += P @ V (V transposed => contiguous B fragments) -----------
#pragma unroll
    for (int t4 = 0; t4 < 4; ++t4) {
      Frag vf = load_frag_B(VtB, Lseq, h * DhC + 16 * t4, k0, lane);
      O[t4] = wmma_bf16(pf, vf, O[t4]);
    }
  }

  // ---- normalize and store attn (B,L,H,Dh) as bf16 ----------------------
#pragma unroll
  for (int j = 0; j < 8; ++j) {
    const float inv = (rsum[j] > 0.f) ? (1.0f / rsum[j]) : 0.f;
    const int m = qt * 16 + j + 8 * hi;
#pragma unroll
    for (int t4 = 0; t4 < 4; ++t4)
      Attnbf[(size_t)(b * Lseq + m) * Dmod + h * DhC + 16 * t4 + n] =
          f32_to_bf16_bits(O[t4][j] * inv);
  }
}

// ---------------------------------------------------------------------------
// Out projection: out = Attnbf (4096x1024) @ Wout -> f32
// ---------------------------------------------------------------------------
__global__ __launch_bounds__(256)
void outproj_kernel(const unsigned short* __restrict__ Attnbf,
                    const unsigned short* __restrict__ WoutT,
                    float* __restrict__ out) {
  const int lane = threadIdx.x & 31;
  const int gw   = blockIdx.x * (blockDim.x >> 5) + (threadIdx.x >> 5);
  const int nW   = Dmod / 64;                // 16
  const int mt   = gw / nW;
  const int nw   = gw - mt * nW;
  const int row0 = mt * 16;
  const int n0   = nw * 64;

  floatx8 acc[4] = {};
  for (int k0 = 0; k0 < Dmod; k0 += 32) {
    Frag a = load_frag_A(Attnbf, Dmod, row0, k0, lane);
    if (k0 + 32 < Dmod) {
      __builtin_prefetch(Attnbf + (size_t)(row0 + (lane & 15)) * Dmod + k0 + 32, 0, 1);
    }
#pragma unroll
    for (int t = 0; t < 4; ++t) {
      Frag b = load_frag_B(WoutT, Dmod, n0 + 16 * t, k0, lane);
      acc[t] = wmma_bf16(a, b, acc[t]);
    }
  }

  const int n = lane & 15, hi = lane >> 4;
#pragma unroll
  for (int j = 0; j < 8; ++j) {
    const int m = row0 + j + 8 * hi;
#pragma unroll
    for (int t = 0; t < 4; ++t)
      out[(size_t)m * Dmod + n0 + 16 * t + n] = acc[t][j];
  }
}

// ---------------------------------------------------------------------------
extern "C" void kernel_launch(void* const* d_in, const int* in_sizes, int n_in,
                              void* d_out, int out_size, void* d_ws, size_t ws_size,
                              hipStream_t stream) {
  (void)in_sizes; (void)n_in; (void)out_size; (void)ws_size;
  const float* x    = (const float*)d_in[0];
  const int*   mask = (const int*)d_in[1];
  const float* Wqkv = (const float*)d_in[2];
  const float* Wout = (const float*)d_in[3];
  float* out = (float*)d_out;

  char* ws = (char*)d_ws;
  const size_t MiB = 1u << 20;
  unsigned short* Xbf    = (unsigned short*)(ws + 0);          //  8 MiB
  unsigned short* WqkvT  = (unsigned short*)(ws + 8  * MiB);   //  6 MiB
  unsigned short* WoutT  = (unsigned short*)(ws + 14 * MiB);   //  2 MiB
  unsigned short* Qbf    = (unsigned short*)(ws + 16 * MiB);   //  8 MiB
  unsigned short* Kbf    = (unsigned short*)(ws + 24 * MiB);   //  8 MiB
  unsigned short* Vt     = (unsigned short*)(ws + 32 * MiB);   //  8 MiB
  unsigned short* Attnbf = (unsigned short*)(ws + 40 * MiB);   //  8 MiB

  // 1) conversions (working set then lives entirely in the 192MB L2)
  {
    int n = Mrows * Dmod;            // 4,194,304
    cvt_f32_bf16<<<(n + 255) / 256, 256, 0, stream>>>(x, Xbf, n);
  }
  {
    int n = Dmod * NQKV;             // 3,145,728
    cvt_f32_bf16_T<<<(n + 255) / 256, 256, 0, stream>>>(Wqkv, WqkvT, Dmod, NQKV);
  }
  {
    int n = Dmod * Dmod;             // 1,048,576
    cvt_f32_bf16_T<<<(n + 255) / 256, 256, 0, stream>>>(Wout, WoutT, Dmod, Dmod);
  }

  // 2) QKV GEMM + RoPE: (4096/16)*(3072/64) = 12288 waves, 8 waves/block
  qkv_rope_kernel<<<12288 / 8, 256, 0, stream>>>(Xbf, WqkvT, Qbf, Kbf, Vt);

  // 3) flash attention: B*H*(L/16) = 4096 waves
  attn_kernel<<<4096 / 8, 256, 0, stream>>>(Qbf, Kbf, Vt, mask, Attnbf);

  // 4) out projection: (4096/16)*(1024/64) = 4096 waves
  outproj_kernel<<<4096 / 8, 256, 0, stream>>>(Attnbf, WoutT, out);
}
